// MultiheadAttention_73693048865382
// MI455X (gfx1250) — compile-verified, matrix operand order checked by reading, not run on hardware
//
#include <hip/hip_runtime.h>

#define S_LEN  2048
#define BATCH  4
#define NHEAD  16
#define DHEAD  64
#define DMODEL 1024

typedef __attribute__((ext_vector_type(16))) __bf16 v16bf;
typedef __attribute__((ext_vector_type(8)))  float  v8f;

union FragA { unsigned u[8]; v16bf v; };

// f32 pair -> packed bf16 dword in exactly one VALU op (gfx1250 native).
__device__ __forceinline__ unsigned pk_bf16(float a, float b) {
  unsigned r;
  asm("v_cvt_pk_bf16_f32 %0, %1, %2" : "=v"(r) : "v"(a), "v"(b));
  return r;
}

// ISA 7.12.2: 16-bit A-matrix 16x32 K-pair index for (vgpr v, lane-half hi)
__device__ __forceinline__ int kpair(int v, int hi) {
  return (v & 3) + hi * 4 + (v >> 2) * 8;
}

// ---------------------------------------------------------------------------
// C = A(MxK) * B(KxN) + bias, fp32 in/out, bf16 WMMA with f32 accumulate.
// 128x128x32 tiles, 256 threads = 8 waves (2x4), 8 WMMAs/wave/K-step.
// Software pipeline: global loads for kt+1 issue BEFORE the WMMAs of kt;
// the cvt+LDS-store (and its s_wait_loadcnt) lands AFTER the compute, so
// HBM latency hides behind the XDL pipe.
// ---------------------------------------------------------------------------
__global__ __launch_bounds__(256)
void gemm_bias_kernel(const float* __restrict__ A, const float* __restrict__ Bm,
                      const float* __restrict__ bias, float* __restrict__ C,
                      int M, int N, int K) {
  __shared__ unsigned lds_a[2][128][16];   // [row][kpair]      (bf16 pairs)
  __shared__ unsigned lds_b[2][128][16];   // [col][kpair] (B transposed)

  const int tid  = threadIdx.x;
  const int lane = tid & 31;
  const int w    = tid >> 5;
  const int hi   = lane >> 4;
  const int lr   = lane & 15;
  const int wm   = w >> 2;                 // 0..1  (64 rows each)
  const int wn   = w & 3;                  // 0..3  (32 cols each)
  const int rowBase = blockIdx.y * 128;
  const int colBase = blockIdx.x * 128;
  const int fr = tid & 7;                  // A: float4 column 0..7
  const int r0 = tid >> 3;                 // A: row 0..31

  const v8f vzero = {0.f,0.f,0.f,0.f,0.f,0.f,0.f,0.f};
  v8f acc[4][2];
  #pragma unroll
  for (int m = 0; m < 4; ++m)
    #pragma unroll
    for (int n = 0; n < 2; ++n) acc[m][n] = vzero;

  const int nk = K / 32;

  // in-flight tile registers
  float4 ta[4];
  float2 tb0[4], tb1[4];

  auto issue_loads = [&](int kt) {
    const int k0 = kt * 32;
    #pragma unroll
    for (int i = 0; i < 4; ++i) {
      const int row = r0 + 32 * i;
      ta[i] = *(const float4*)(A + (size_t)(rowBase + row) * K + k0 + fr * 4);
    }
    #pragma unroll
    for (int i = 0; i < 4; ++i) {
      const int c  = tid + 256 * i;        // (kpair, npair) cell
      const int kp = c >> 6;
      const int np = c & 63;
      const int k  = k0 + kp * 2;
      const int n  = colBase + np * 2;
      tb0[i] = *(const float2*)(Bm + (size_t)k * N + n);
      tb1[i] = *(const float2*)(Bm + (size_t)(k + 1) * N + n);
    }
  };
  auto cvt_store = [&](int buf) {
    #pragma unroll
    for (int i = 0; i < 4; ++i) {
      const int row = r0 + 32 * i;
      lds_a[buf][row][fr * 2]     = pk_bf16(ta[i].x, ta[i].y);
      lds_a[buf][row][fr * 2 + 1] = pk_bf16(ta[i].z, ta[i].w);
    }
    #pragma unroll
    for (int i = 0; i < 4; ++i) {
      const int c  = tid + 256 * i;
      const int kp = c >> 6;
      const int np = c & 63;
      lds_b[buf][np * 2][kp]     = pk_bf16(tb0[i].x, tb1[i].x);
      lds_b[buf][np * 2 + 1][kp] = pk_bf16(tb0[i].y, tb1[i].y);
    }
  };

  issue_loads(0);
  cvt_store(0);
  __syncthreads();
  int buf = 0;
  for (int kt = 0; kt < nk; ++kt) {
    const bool more = (kt + 1 < nk);
    if (more) issue_loads(kt + 1);       // global loads in flight during WMMAs

    FragA fa[4], fb[2];
    #pragma unroll
    for (int m = 0; m < 4; ++m)
      #pragma unroll
      for (int v = 0; v < 8; ++v)
        fa[m].u[v] = lds_a[buf][wm * 64 + m * 16 + lr][kpair(v, hi)];
    #pragma unroll
    for (int n = 0; n < 2; ++n)
      #pragma unroll
      for (int v = 0; v < 8; ++v)
        fb[n].u[v] = lds_b[buf][wn * 32 + n * 16 + lr][kpair(v, hi)];
    #pragma unroll
    for (int m = 0; m < 4; ++m)
      #pragma unroll
      for (int n = 0; n < 2; ++n)
        acc[m][n] = __builtin_amdgcn_wmma_f32_16x16x32_bf16(
            false, fa[m].v, false, fb[n].v, (short)0, acc[m][n], false, false);

    if (more) cvt_store(buf ^ 1);        // consume loads after the compute
    __syncthreads();
    buf ^= 1;
  }

  // epilogue: bias add + store (C layout: vgpr i -> row i+8*hi, lane lr -> col)
  #pragma unroll
  for (int n = 0; n < 2; ++n) {
    const int gc = colBase + wn * 32 + n * 16 + lr;
    const float bv = bias[gc];
    #pragma unroll
    for (int m = 0; m < 4; ++m)
      #pragma unroll
      for (int i = 0; i < 8; ++i) {
        const int gr = rowBase + wm * 64 + m * 16 + i + 8 * hi;
        C[(size_t)gr * N + gc] = acc[m][n][i] + bv;
      }
  }
}

// ---------------------------------------------------------------------------
// Flash-attention style: per (b,h), 128-query blocks (16 q-rows per wave),
// stream 32-key tiles; online softmax in f32; QK^T and PV via bf16 WMMA.
// Double-buffered K/V LDS tiles with the same loads-early pipeline.
// Reference quirk preserved: scale = 1/sqrt(H) = 0.25, no masking.
// ---------------------------------------------------------------------------
__global__ __launch_bounds__(256)
void attn_kernel(const float* __restrict__ Q, const float* __restrict__ Kp,
                 const float* __restrict__ V, float* __restrict__ O) {
  __shared__ unsigned lds_k[2][32][32];              // [key][d-pair]
  __shared__ unsigned lds_v[2][64][16];              // [d][key-pair] (transposed)
  __shared__ __align__(4) __bf16 lds_p[8][16][32];   // per-wave P bounce

  const int tid  = threadIdx.x;
  const int lane = tid & 31;
  const int w    = tid >> 5;
  const int hi   = lane >> 4;
  const int lr   = lane & 15;
  const int h    = blockIdx.y;
  const int b    = blockIdx.z;
  const int qBase = blockIdx.x * 128 + w * 16;
  const size_t rowOff = (size_t)b * S_LEN;
  const int hOff = h * DHEAD;
  const float scale = 0.25f;                          // 1/sqrt(NHEAD)

  // Q A-fragments (held in registers for all key tiles), d-groups 0..1
  FragA qf[2];
  {
    const float* qrow = Q + (rowOff + qBase + lr) * DMODEL + hOff;
    #pragma unroll
    for (int g = 0; g < 2; ++g)
      #pragma unroll
      for (int v = 0; v < 8; ++v) {
        const int d = 2 * (g * 16 + kpair(v, hi));
        const float2 q2 = *(const float2*)(qrow + d);
        qf[g].u[v] = pk_bf16(q2.x, q2.y);
      }
  }

  const v8f vzero = {0.f,0.f,0.f,0.f,0.f,0.f,0.f,0.f};
  v8f o[4];
  #pragma unroll
  for (int n = 0; n < 4; ++n) o[n] = vzero;
  float m_i[8], l_i[8];
  #pragma unroll
  for (int i = 0; i < 8; ++i) { m_i[i] = -3.0e38f; l_i[i] = 0.f; }

  // in-flight K/V tile registers
  float4 tk[2];
  float  tv0[4], tv1[4];

  auto issue_kv = [&](int kt) {
    #pragma unroll
    for (int i = 0; i < 2; ++i) {
      const int c   = tid + 256 * i;       // 0..511
      const int key = c >> 4;              // 0..31
      const int f4  = c & 15;              // 0..15
      tk[i] = *(const float4*)(Kp + (rowOff + kt * 32 + key) * DMODEL + hOff + f4 * 4);
    }
    #pragma unroll
    for (int i = 0; i < 4; ++i) {
      const int c  = tid + 256 * i;        // 0..1023
      const int kp = c >> 6;               // 0..15
      const int d  = c & 63;
      const float* vp = V + (rowOff + kt * 32 + kp * 2) * DMODEL + hOff + d;
      tv0[i] = vp[0];
      tv1[i] = vp[DMODEL];
    }
  };
  auto store_kv = [&](int buf) {
    #pragma unroll
    for (int i = 0; i < 2; ++i) {
      const int c   = tid + 256 * i;
      const int key = c >> 4;
      const int f4  = c & 15;
      lds_k[buf][key][f4 * 2]     = pk_bf16(tk[i].x, tk[i].y);
      lds_k[buf][key][f4 * 2 + 1] = pk_bf16(tk[i].z, tk[i].w);
    }
    #pragma unroll
    for (int i = 0; i < 4; ++i) {
      const int c  = tid + 256 * i;
      const int kp = c >> 6;
      const int d  = c & 63;
      lds_v[buf][d][kp] = pk_bf16(tv0[i], tv1[i]);
    }
  };

  issue_kv(0);
  store_kv(0);
  __syncthreads();
  int buf = 0;

  for (int kt = 0; kt < S_LEN / 32; ++kt) {
    const bool more = (kt + 1 < S_LEN / 32);
    if (more) issue_kv(kt + 1);          // global loads in flight during compute

    // scores: 2 key-column groups x 2 d-groups -> 16x32 f32
    v8f sc[2];
    #pragma unroll
    for (int g = 0; g < 2; ++g) {
      v8f s = vzero;
      #pragma unroll
      for (int dg = 0; dg < 2; ++dg) {
        FragA kf;
        #pragma unroll
        for (int v = 0; v < 8; ++v)
          kf.u[v] = lds_k[buf][g * 16 + lr][dg * 16 + kpair(v, hi)];
        s = __builtin_amdgcn_wmma_f32_16x16x32_bf16(
                false, qf[dg].v, false, kf.v, (short)0, s, false, false);
      }
      sc[g] = s;
    }

    // online softmax (row = vgpr i + 8*hi, 16 cols per lane-half per group)
    float alpha[8];
    #pragma unroll
    for (int i = 0; i < 8; ++i) {
      float s0 = sc[0][i] * scale;
      float s1 = sc[1][i] * scale;
      float t = fmaxf(s0, s1);
      #pragma unroll
      for (int d = 1; d < 16; d <<= 1) t = fmaxf(t, __shfl_xor(t, d, 32));
      const float mn = fmaxf(m_i[i], t);
      alpha[i] = __expf(m_i[i] - mn);
      m_i[i] = mn;
      const float p0 = __expf(s0 - mn);
      const float p1 = __expf(s1 - mn);
      sc[0][i] = p0; sc[1][i] = p1;
      float rs = p0 + p1;
      #pragma unroll
      for (int d = 1; d < 16; d <<= 1) rs += __shfl_xor(rs, d, 32);
      l_i[i] = l_i[i] * alpha[i] + rs;
    }
    #pragma unroll
    for (int n = 0; n < 4; ++n)
      #pragma unroll
      for (int i = 0; i < 8; ++i) o[n][i] *= alpha[i];

    // P: C-layout f32 -> A-layout bf16 via wave-private LDS bounce
    #pragma unroll
    for (int g = 0; g < 2; ++g)
      #pragma unroll
      for (int i = 0; i < 8; ++i)
        lds_p[w][i + 8 * hi][g * 16 + lr] = (__bf16)sc[g][i];
    asm volatile("s_wait_dscnt 0" ::: "memory");   // wave-internal LDS ordering
    FragA pf;
    #pragma unroll
    for (int v = 0; v < 8; ++v)
      pf.u[v] = *(const unsigned*)&lds_p[w][lr][2 * kpair(v, hi)];

    // O += P @ V  (4 d-column groups)
    #pragma unroll
    for (int n = 0; n < 4; ++n) {
      FragA vf;
      #pragma unroll
      for (int v = 0; v < 8; ++v)
        vf.u[v] = lds_v[buf][n * 16 + lr][kpair(v, hi)];
      o[n] = __builtin_amdgcn_wmma_f32_16x16x32_bf16(
                 false, pf.v, false, vf.v, (short)0, o[n], false, false);
    }

    if (more) store_kv(buf ^ 1);         // consume loads after the compute
    __syncthreads();
    buf ^= 1;
  }

  // finalize: divide by row sums, write (b, s, h*dv) fp32
  #pragma unroll
  for (int i = 0; i < 8; ++i) l_i[i] = 1.f / l_i[i];
  #pragma unroll
  for (int n = 0; n < 4; ++n)
    #pragma unroll
    for (int i = 0; i < 8; ++i) {
      const int gq = qBase + i + 8 * hi;
      O[(rowOff + gq) * DMODEL + hOff + n * 16 + lr] = o[n][i] * l_i[i];
    }
}

// ---------------------------------------------------------------------------
extern "C" void kernel_launch(void* const* d_in, const int* in_sizes, int n_in,
                              void* d_out, int out_size, void* d_ws, size_t ws_size,
                              hipStream_t stream) {
  (void)in_sizes; (void)n_in; (void)out_size; (void)ws_size;
  const float* x  = (const float*)d_in[0];
  const float* wq = (const float*)d_in[1];
  const float* bq = (const float*)d_in[2];
  const float* wk = (const float*)d_in[3];
  const float* bk = (const float*)d_in[4];
  const float* wv = (const float*)d_in[5];
  const float* bv = (const float*)d_in[6];
  const float* wo = (const float*)d_in[7];
  const float* bo = (const float*)d_in[8];
  float* out = (float*)d_out;

  const int M = BATCH * S_LEN;   // 8192
  const int N = DMODEL;          // 1024
  const int K = DMODEL;          // 1024

  float* ws = (float*)d_ws;
  const size_t plane = (size_t)M * N;      // 8.4M floats = 33.5 MB
  float* q = ws;
  float* k = ws + plane;
  float* v = ws + 2 * plane;
  float* a = ws + 3 * plane;

  dim3 blk(256);
  dim3 ggrd(N / 128, M / 128);             // (8, 64)
  gemm_bias_kernel<<<ggrd, blk, 0, stream>>>(x, wq, bq, q, M, N, K);
  gemm_bias_kernel<<<ggrd, blk, 0, stream>>>(x, wk, bk, k, M, N, K);
  gemm_bias_kernel<<<ggrd, blk, 0, stream>>>(x, wv, bv, v, M, N, K);

  dim3 agrd(S_LEN / 128, NHEAD, BATCH);    // (16, 16, 4)
  attn_kernel<<<agrd, blk, 0, stream>>>(q, k, v, a);

  gemm_bias_kernel<<<ggrd, blk, 0, stream>>>(a, wo, bo, out, M, N, K);
}
